// ZigzagRingFlashAttention_85323820302505
// MI455X (gfx1250) — matrix operand order repaired and checked by reading, not compiled, not run on hardware
//
#include <hip/hip_runtime.h>
#include <math.h>

typedef __attribute__((ext_vector_type(16))) _Float16 v16h;
typedef __attribute__((ext_vector_type(8)))  float    v8f;

#define S_LEN 4096
#define NH    8
#define HD    64
#define WSZ   4           // simulated ring world size
#define CHK   512         // S / (2*WSZ)

// D = A*B + C ; A 16x32 f16, B 32x16 f16, C/D 16x16 f32
__device__ __forceinline__ v8f wmma_f16(v16h a, v16h b, v8f c) {
    return __builtin_amdgcn_wmma_f32_16x16x32_f16(
        /*neg_a=*/false, a, /*neg_b=*/false, b,
        /*c_mod=*/(short)0, c, /*reuse_a=*/false, /*reuse_b=*/false);
}

// Block = 4 wave32s, one 64-query super-tile, K/V tiles shared via LDS in
// pre-converted, fragment-ordered f16 layout. Double-buffered pipeline.
__global__ __launch_bounds__(128)
void zigzag_flash_fwd(const float* __restrict__ Q,
                      const float* __restrict__ K,
                      const float* __restrict__ V,
                      float* __restrict__ Out,   // [W,1,2C,H,D]
                      float* __restrict__ Lse)   // [W,1,H,2C]
{
    // Fragment-ordered tiles: 4 frags * 32 lanes * 16 f16 = 2048 f16 = 4KB each
    __shared__ __align__(16) _Float16 smemK[2][2048];
    __shared__ __align__(16) _Float16 smemV[2][2048];

    const int tid  = threadIdx.x;
    const int lane = tid & 31;
    const int wave = tid >> 5;
    const int head = blockIdx.y;
    const int bq0  = blockIdx.x * 64;         // block's first query row
    const int q0   = bq0 + wave * 16;         // this wave's query tile
    const int n16  = lane & 15;
    const int hh   = lane >> 4;

    // softmax kept in exp2 domain; scale folded into Q fragment
    const float SCL = 0.125f * 1.4426950408889634f;   // 1/sqrt(D) * log2(e)
    const float LN2 = 0.6931471805599453f;

    // staging role: one fragment-slot (32 bytes of one frag) per thread
    const int fid = tid >> 5;                 // frag id 0..3
    const int Ls  = tid & 31;                 // slot lane
    const int sm  = Ls & 15;
    const int shh = Ls >> 4;
    const int ksub = fid & 1, kdc = fid >> 1; // K frag f = sub + 2*dc

    // ---------------- persistent Q^T B-fragments (pre-scaled) ----------------
    v16h bq[2];
    {
        const float* qrow = Q + ((size_t)(q0 + n16) * NH + head) * HD;
        #pragma unroll
        for (int dc = 0; dc < 2; ++dc) {
            const int base = dc * 32 + 16 * hh;
            #pragma unroll
            for (int i = 0; i < 4; ++i) {
                float4 f = *(const float4*)(qrow + base + 4 * i);
                bq[dc][4*i+0] = (_Float16)(f.x * SCL);
                bq[dc][4*i+1] = (_Float16)(f.y * SCL);
                bq[dc][4*i+2] = (_Float16)(f.z * SCL);
                bq[dc][4*i+3] = (_Float16)(f.w * SCL);
            }
        }
    }

    v8f o[4];
    #pragma unroll
    for (int t = 0; t < 4; ++t)
        #pragma unroll
        for (int r = 0; r < 8; ++r) o[t][r] = 0.f;
    float m_run = -1e30f, l_run = 0.f;

    const int nkt = (bq0 + 63) / 32 + 1;      // block causal bound (uniform)

    float2 kr[8];
    float  vr[16];
    {   // prologue: stage tile 0
        const float* krow = K + ((size_t)(16 * ksub + sm) * NH + head) * HD;
        #pragma unroll
        for (int d = 0; d < 8; ++d) {
            const int g = d >> 2, j = d & 3;
            kr[d] = *(const float2*)(krow + kdc * 32 + 8 * g + 16 * shh + 2 * j);
        }
        const float* vcol = V + ((size_t)(8 * shh) * NH + head) * HD + 16 * fid + sm;
        #pragma unroll
        for (int e = 0; e < 16; ++e) vr[e] = vcol[(size_t)(e + 8 * (e >> 3)) * NH * HD];

        v16h kf, vf;
        #pragma unroll
        for (int d = 0; d < 8; ++d) { kf[2*d] = (_Float16)kr[d].x; kf[2*d+1] = (_Float16)kr[d].y; }
        #pragma unroll
        for (int e = 0; e < 16; ++e) vf[e] = (_Float16)vr[e];
        *(v16h*)&smemK[0][fid * 512 + Ls * 16] = kf;
        *(v16h*)&smemV[0][fid * 512 + Ls * 16] = vf;
    }
    __syncthreads();

    for (int kt = 0; kt < nkt; ++kt) {
        const int kbase = kt * 32;
        const int b = kt & 1;
        const bool have_next = (kt + 1) < nkt;

        // -------- kick off next tile's global loads (latency hidden) --------
        if (have_next) {
            const int nb = kbase + 32;
            const float* krow = K + ((size_t)(nb + 16 * ksub + sm) * NH + head) * HD;
            #pragma unroll
            for (int d = 0; d < 8; ++d) {
                const int g = d >> 2, j = d & 3;
                kr[d] = *(const float2*)(krow + kdc * 32 + 8 * g + 16 * shh + 2 * j);
            }
            const float* vcol = V + ((size_t)(nb + 8 * shh) * NH + head) * HD + 16 * fid + sm;
            #pragma unroll
            for (int e = 0; e < 16; ++e) vr[e] = vcol[(size_t)(e + 8 * (e >> 3)) * NH * HD];
        }

        // -------- compute current tile from LDS (wave-uniform predicate) ----
        if (kbase <= q0 + 15) {
            v16h ak0 = *(const v16h*)&smemK[b][0 * 512 + lane * 16]; // sub0,dc0
            v16h ak1 = *(const v16h*)&smemK[b][1 * 512 + lane * 16]; // sub1,dc0
            v16h ak2 = *(const v16h*)&smemK[b][2 * 512 + lane * 16]; // sub0,dc1
            v16h ak3 = *(const v16h*)&smemK[b][3 * 512 + lane * 16]; // sub1,dc1

            // S^T = K * Q^T (already exp2-domain scaled via bq)
            v8f c0, c1;
            #pragma unroll
            for (int r = 0; r < 8; ++r) { c0[r] = 0.f; c1[r] = 0.f; }
            c0 = wmma_f16(ak0, bq[0], c0);
            c0 = wmma_f16(ak2, bq[1], c0);
            c1 = wmma_f16(ak1, bq[0], c1);
            c1 = wmma_f16(ak3, bq[1], c1);

            float sv[16];
            #pragma unroll
            for (int r = 0; r < 8; ++r) { sv[r] = c0[r]; sv[8 + r] = c1[r]; }

            // causal mask: only the 1-2 diagonal tiles pay (uniform branch);
            // local key of sv[i] is 16*(i/8) + (i%8) + 8*hh; mask if > qdiag
            if (kbase + 31 > q0) {
                const int qdiag = q0 + n16 - 8 * hh - kbase;
                #pragma unroll
                for (int i = 0; i < 16; ++i) {
                    const int kloc = 16 * (i >> 3) + (i & 7);  // compile-time imm
                    if (kloc > qdiag) sv[i] = -1e30f;
                }
            }

            // online softmax stats (halves merged via xor16)
            float mt = sv[0];
            #pragma unroll
            for (int i = 1; i < 16; ++i) mt = fmaxf(mt, sv[i]);
            mt = fmaxf(mt, __shfl_xor(mt, 16));
            const float m_new = fmaxf(m_run, mt);

            // rescale only when some row max actually moved (rare after warmup)
            if (__any(m_new > m_run)) {
                const float corr = __builtin_amdgcn_exp2f(m_run - m_new);
                l_run *= corr;
                #pragma unroll
                for (int r = 0; r < 8; ++r) {
                    const float fr = __shfl(corr, r + 8 * hh);
                    #pragma unroll
                    for (int t = 0; t < 4; ++t) o[t][r] *= fr;
                }
                m_run = m_new;
            }

            float pv[16];
            float ls = 0.f;
            #pragma unroll
            for (int i = 0; i < 16; ++i) {
                pv[i] = __builtin_amdgcn_exp2f(sv[i] - m_run);
                ls += pv[i];
            }
            ls += __shfl_xor(ls, 16);
            l_run += ls;

            // pack P into A-frag (pure per-lane: layouts line up)
            v16h pa;
            #pragma unroll
            for (int d = 0; d < 8; ++d) {
                const int g = d >> 2, j = d & 3;
                pa[2*d]   = (_Float16)pv[g * 8 + 2*j];
                pa[2*d+1] = (_Float16)pv[g * 8 + 2*j + 1];
            }

            // V B-frags from LDS + accumulate O
            #pragma unroll
            for (int t = 0; t < 4; ++t) {
                v16h bv = *(const v16h*)&smemV[b][t * 512 + lane * 16];
                o[t] = wmma_f16(pa, bv, o[t]);
            }
        }

        // -------- convert + store staged tile into other buffer -------------
        if (have_next) {
            v16h kf, vf;
            #pragma unroll
            for (int d = 0; d < 8; ++d) { kf[2*d] = (_Float16)kr[d].x; kf[2*d+1] = (_Float16)kr[d].y; }
            #pragma unroll
            for (int e = 0; e < 16; ++e) vf[e] = (_Float16)vr[e];
            *(v16h*)&smemK[b ^ 1][fid * 512 + Ls * 16] = kf;
            *(v16h*)&smemV[b ^ 1][fid * 512 + Ls * 16] = vf;
        }
        __syncthreads();
    }

    // ---------------- epilogue: 1/l per row, zigzag scatter ----------------
    float linv[8];
    #pragma unroll
    for (int r = 0; r < 8; ++r)
        linv[r] = 1.f / __shfl(l_run, r + 8 * hh);

    #pragma unroll
    for (int r = 0; r < 8; ++r) {
        const int s     = q0 + r + 8 * hh;
        const int chunk = s >> 9;
        const int co    = s & (CHK - 1);
        const int rank  = (chunk < WSZ) ? chunk : (2 * WSZ - 1 - chunk);
        const int slot  = (chunk < WSZ) ? 0 : 1;
        const size_t row = (size_t)rank * (2 * CHK) + slot * CHK + co;
        #pragma unroll
        for (int t = 0; t < 4; ++t) {
            const int col = 16 * t + n16;
            Out[(row * NH + head) * HD + col] = o[t][r] * linv[r];
        }
    }

    if (lane < 16) {
        const int s     = q0 + lane;
        const int chunk = s >> 9;
        const int co    = s & (CHK - 1);
        const int rank  = (chunk < WSZ) ? chunk : (2 * WSZ - 1 - chunk);
        const int slot  = (chunk < WSZ) ? 0 : 1;
        Lse[(size_t)rank * NH * (2 * CHK) + (size_t)head * (2 * CHK) + slot * CHK + co]
            = LN2 * (m_run + __builtin_amdgcn_logf(l_run));
    }
}

extern "C" void kernel_launch(void* const* d_in, const int* in_sizes, int n_in,
                              void* d_out, int out_size, void* d_ws, size_t ws_size,
                              hipStream_t stream) {
    const float* q = (const float*)d_in[0];
    const float* k = (const float*)d_in[1];
    const float* v = (const float*)d_in[2];
    float* out = (float*)d_out;                            // [4,1,1024,8,64]
    float* lse = out + (size_t)WSZ * (2 * CHK) * NH * HD;  // [4,1,8,1024]

    dim3 grid(S_LEN / 64, NH);   // (64, 8): 4 waves/block, 64 queries/block
    dim3 block(128);
    zigzag_flash_fwd<<<grid, block, 0, stream>>>(q, k, v, out, lse);
}
